// GCN0_2456721293643
// MI455X (gfx1250) — matrix operand order repaired and checked by reading, not compiled, not run on hardware
//
#include <hip/hip_runtime.h>

#define NN 100000
#define NE 1600000
#define FF 128
#define OUTF 64
#define MTILES 6250   // 100000 / 16

typedef __attribute__((ext_vector_type(16))) __bf16 v16bf;
typedef __attribute__((ext_vector_type(8)))  __bf16 v8bf;
typedef __attribute__((ext_vector_type(4)))  __bf16 v4bf;
typedef __attribute__((ext_vector_type(8)))  float  v8f;

// ---------------- zero accumulators ----------------
__global__ void k_zero(float* __restrict__ p, long n) {
    long i = ((long)blockIdx.x * blockDim.x + threadIdx.x) * 4;
    if (i < n) *(float4*)(p + i) = make_float4(0.f, 0.f, 0.f, 0.f);
}

// ---------------- degree histogram (f32 atomic adds, exact for counts < 2^24) ----
__global__ void k_deg(const int* __restrict__ src, const int* __restrict__ dst,
                      float* __restrict__ dout, float* __restrict__ din, int ne) {
    int e = blockIdx.x * blockDim.x + threadIdx.x;
    if (e < ne) {
        unsafeAtomicAdd(&dout[src[e]], 1.0f);
        unsafeAtomicAdd(&din[dst[e]], 1.0f);
    }
}

// ---------------- normalization factors ----------------
__global__ void k_inv(const float* __restrict__ dout, const float* __restrict__ din,
                      float* __restrict__ iso, float* __restrict__ isi,
                      float* __restrict__ idg, int n) {
    int i = blockIdx.x * blockDim.x + threadIdx.x;
    if (i < n) {
        float o = dout[i], d = din[i];
        iso[i] = (o > 0.f) ? rsqrtf(fmaxf(o, 1.f)) : 0.f;
        isi[i] = (d > 0.f) ? rsqrtf(fmaxf(d, 1.f)) : 0.f;
        idg[i] = 1.f / fmaxf(d, 1.f);
    }
}

// ---------------- pack weight matrix [K=128][ncols] into B-fragment layout ------
// B (32x16 per 16x16x32 wmma): lanes 0-15 hold K = kt*32 + {0..7, 16..23}, lanes 16-31 hold
// K = kt*32 + {8..15, 24..31}; element j is K = base + (j&7) + (j>>3)*16.
// Packed addr: Bp[(((nt*4 + kt)*32 + lane)*16) + j]  -> contiguous 32B per lane.
__global__ void k_packB(const float* __restrict__ W, __bf16* __restrict__ Bp, int ncols) {
    int t = blockIdx.x * blockDim.x + threadIdx.x;
    int total = ncols * 128;           // (ncols/16)*4*32*16
    if (t >= total) return;
    int j    = t & 15;
    int lane = (t >> 4) & 31;
    int kt   = (t >> 9) & 3;
    int nt   = t >> 11;
    int n = nt * 16 + (lane & 15);
    int K = kt * 32 + ((lane >> 4) * 8) + (j & 7) + ((j >> 3) * 16);
    Bp[t] = (__bf16)W[K * ncols + n];
}

// ---------------- edge scatter: out[dst] += rows[src] * (scale[src]?) ----------
// one wave per edge: float4 per lane covers 128 feats; 4 global f32 atomics/lane.
// node tables fit in 192MB L2, so gathers + atomic RMWs are L2-local.
template <bool SCALE>
__global__ void k_scatter(const float* __restrict__ rows, const int* __restrict__ src,
                          const int* __restrict__ dst, const float* __restrict__ sc,
                          float* __restrict__ out, int ne) {
    int lane = threadIdx.x & 31;
    int e = blockIdx.x * 8 + (threadIdx.x >> 5);
    if (e >= ne) return;
    int s = src[e], d = dst[e];
    float m = SCALE ? sc[s] : 1.0f;
    float4 v = ((const float4*)(rows + (size_t)s * FF))[lane];
    float* o = out + (size_t)d * FF + lane * 4;
    unsafeAtomicAdd(o + 0, v.x * m);
    unsafeAtomicAdd(o + 1, v.y * m);
    unsafeAtomicAdd(o + 2, v.z * m);
    unsafeAtomicAdd(o + 3, v.w * m);
}

// ---------------- row-scale + f32 -> bf16 convert (A matrix producer) ----------
__global__ void k_scale_cvt(const float* __restrict__ in, const float* __restrict__ rscale,
                            __bf16* __restrict__ out, int nrows) {
    int t = blockIdx.x * blockDim.x + threadIdx.x;   // one per 4 elems
    int total = nrows * (FF / 4);
    if (t >= total) return;
    int row = t >> 5;           // FF/4 == 32
    int c4  = t & 31;
    float s = rscale[row];
    float4 v = ((const float4*)(in + (size_t)row * FF))[c4];
    v4bf o;
    o.x = (__bf16)(v.x * s);
    o.y = (__bf16)(v.y * s);
    o.z = (__bf16)(v.z * s);
    o.w = (__bf16)(v.w * s);
    ((v4bf*)(out + (size_t)row * FF))[c4] = o;
}

// ---- async stage of packed weights into LDS: 256 threads x 8 x 16B = 32KB ----
// GLOBAL_LOAD_ASYNC_TO_LDS_B128: per-lane LDS byte offset in a VGPR (low 32 bits
// of the generic shared pointer == LDS offset), per-lane 64-bit global address.
// Tracked by ASYNCcnt; complete with s_wait_asynccnt 0 then barrier.
__device__ __forceinline__ void stage32k(const __bf16* __restrict__ gsrc, __bf16* lds) {
    int tid = threadIdx.x;
    unsigned lbase = (unsigned)(size_t)lds;
#pragma unroll
    for (int i = 0; i < 8; ++i) {
        unsigned off = (unsigned)(tid * 16 + i * 4096);
        unsigned l = lbase + off;
        const char* g = (const char*)gsrc + off;
        asm volatile("global_load_async_to_lds_b128 %0, %1, off"
                     :: "v"(l), "v"(g) : "memory");
    }
    asm volatile("s_wait_asynccnt 0" ::: "memory");
    __syncthreads();
}

// A fragment (16x32 bf16, row-major source, ld=128):
// lane<16: row M=lane, K = kt*32 + {0..7, 16..23}; lane>=16: M=lane-16, K += 8.
__device__ __forceinline__ v16bf load_a(const __bf16* __restrict__ arow, int kt) {
    v8bf lo = *(const v8bf*)(arow + kt * 32);
    v8bf hi = *(const v8bf*)(arow + kt * 32 + 16);
    return __builtin_shufflevector(lo, hi, 0,1,2,3,4,5,6,7,8,9,10,11,12,13,14,15);
}

// ---------------- GEMM1: h = relu(A(100000x128,bf16) @ W1 + b1) ---------------
// B lives in LDS (shared by all 8 waves); inner loop is ds_load + wmma.
// writes h as f32 (for scatter2) and bf16 (for GEMM2)
__global__ void k_gemm1(const __bf16* __restrict__ Ab, const __bf16* __restrict__ Bp,
                        const float* __restrict__ bias, float* __restrict__ Hf,
                        __bf16* __restrict__ Hb, int mtiles) {
    __shared__ alignas(32) __bf16 Bs[FF * FF];   // 32 KB packed W1
    stage32k(Bp, Bs);

    int lane = threadIdx.x & 31;
    int mt = blockIdx.x * 8 + (threadIdx.x >> 5);
    if (mt >= mtiles) return;
    int m0 = mt * 16;
    const __bf16* arow = Ab + (size_t)(m0 + (lane & 15)) * FF + ((lane >> 4) * 8);

    v8f acc[8] = {};
#pragma unroll
    for (int kt = 0; kt < 4; ++kt) {
        v16bf a = load_a(arow, kt);
#pragma unroll
        for (int nt = 0; nt < 8; ++nt) {
            v16bf b = *(const v16bf*)(Bs + (((nt * 4 + kt) * 32 + lane) << 4));
            acc[nt] = __builtin_amdgcn_wmma_f32_16x16x32_bf16(
                false, a, false, b, (short)0, acc[nt], false, false);
        }
    }
    int ncol  = lane & 15;
    int rbase = m0 + ((lane >> 4) << 3);   // +8 rows for upper half-wave
#pragma unroll
    for (int nt = 0; nt < 8; ++nt) {
        int n = nt * 16 + ncol;
        float bv = bias[n];
#pragma unroll
        for (int r = 0; r < 8; ++r) {
            float v = fmaxf(acc[nt][r] + bv, 0.f);
            size_t idx = (size_t)(rbase + r) * FF + n;
            Hf[idx] = v;
            Hb[idx] = (__bf16)v;
        }
    }
}

// ---------------- GEMM2: out = Hb @ Wself + NMb @ Wneigh + b2 ------------------
// BpS and BpN are contiguous in the workspace -> single 32KB async stage.
__global__ void k_gemm2(const __bf16* __restrict__ Hb, const __bf16* __restrict__ NMb,
                        const __bf16* __restrict__ BpS,
                        const float* __restrict__ bias, float* __restrict__ out, int mtiles) {
    __shared__ alignas(32) __bf16 Bs[2 * FF * OUTF];   // [0:8192)=Wself, [8192:16384)=Wneigh
    stage32k(BpS, Bs);

    int lane = threadIdx.x & 31;
    int mt = blockIdx.x * 8 + (threadIdx.x >> 5);
    if (mt >= mtiles) return;
    int m0 = mt * 16;
    size_t aoff = (size_t)(m0 + (lane & 15)) * FF + ((lane >> 4) * 8);
    const __bf16* a1row = Hb + aoff;
    const __bf16* a2row = NMb + aoff;

    v8f acc[4] = {};
#pragma unroll
    for (int kt = 0; kt < 4; ++kt) {
        v16bf a1 = load_a(a1row, kt);
        v16bf a2 = load_a(a2row, kt);
#pragma unroll
        for (int nt = 0; nt < 4; ++nt) {
            int foff = ((nt * 4 + kt) * 32 + lane) << 4;
            v16bf bs = *(const v16bf*)(Bs + foff);
            acc[nt] = __builtin_amdgcn_wmma_f32_16x16x32_bf16(
                false, a1, false, bs, (short)0, acc[nt], false, false);
            v16bf bn = *(const v16bf*)(Bs + FF * OUTF + foff);
            acc[nt] = __builtin_amdgcn_wmma_f32_16x16x32_bf16(
                false, a2, false, bn, (short)0, acc[nt], false, false);
        }
    }
    int ncol  = lane & 15;
    int rbase = m0 + ((lane >> 4) << 3);
#pragma unroll
    for (int nt = 0; nt < 4; ++nt) {
        int n = nt * 16 + ncol;
        float bv = bias[n];
#pragma unroll
        for (int r = 0; r < 8; ++r)
            out[(size_t)(rbase + r) * OUTF + n] = acc[nt][r] + bv;
    }
}

extern "C" void kernel_launch(void* const* d_in, const int* in_sizes, int n_in,
                              void* d_out, int out_size, void* d_ws, size_t ws_size,
                              hipStream_t stream) {
    const float* x  = (const float*)d_in[0];
    const float* W1 = (const float*)d_in[1];
    const float* b1 = (const float*)d_in[2];
    const float* Ws = (const float*)d_in[3];
    const float* Wn = (const float*)d_in[4];
    const float* b2 = (const float*)d_in[5];
    const int*  src = (const int*)d_in[6];
    const int*  dst = (const int*)d_in[7];
    float* out = (float*)d_out;

    const size_t NPAD = 100032;            // 100000 rounded to 64 (256B-aligned regions)
    const size_t NF   = (size_t)NN * FF;   // 12,800,000

    float* deg_out = (float*)d_ws;
    float* deg_in  = deg_out + NPAD;
    float* agg     = deg_in + NPAD;        // zeroed region: deg_out,deg_in,agg,neigh
    float* neigh   = agg + NF;
    float* inv_out = neigh + NF;
    float* inv_in  = inv_out + NPAD;
    float* inv_dg  = inv_in + NPAD;
    float* Hf      = inv_dg + NPAD;
    __bf16* Hb  = (__bf16*)(Hf + NF);
    __bf16* Ab  = Hb + NF;
    __bf16* NMb = Ab + NF;
    __bf16* W1p = NMb + NF;
    __bf16* Wsp = W1p + (size_t)FF * FF;
    __bf16* Wnp = Wsp + (size_t)FF * OUTF;   // contiguous after Wsp (gemm2 stages both)
    size_t need = (size_t)((char*)(Wnp + (size_t)FF * OUTF) - (char*)d_ws);
    if (ws_size < need) return;

    long zcount = (long)(2 * NPAD + 2 * NF);
    k_zero<<<(unsigned)((zcount / 4 + 255) / 256), 256, 0, stream>>>(deg_out, zcount);

    k_deg<<<(NE + 255) / 256, 256, 0, stream>>>(src, dst, deg_out, deg_in, NE);
    k_inv<<<(NN + 255) / 256, 256, 0, stream>>>(deg_out, deg_in, inv_out, inv_in, inv_dg, NN);

    k_packB<<<(FF * FF) / 256, 256, 0, stream>>>(W1, W1p, FF);
    k_packB<<<(FF * OUTF) / 256, 256, 0, stream>>>(Ws, Wsp, OUTF);
    k_packB<<<(FF * OUTF) / 256, 256, 0, stream>>>(Wn, Wnp, OUTF);

    // GraphConv aggregate: agg[dst] += x[src] * inv_sqrt_out[src]
    k_scatter<true><<<NE / 8, 256, 0, stream>>>(x, src, dst, inv_out, agg, NE);
    // A = bf16(agg * inv_sqrt_in)
    k_scale_cvt<<<(NN * 32 + 255) / 256, 256, 0, stream>>>(agg, inv_in, Ab, NN);
    // h = relu(A @ W1 + b1)
    k_gemm1<<<(MTILES + 7) / 8, 256, 0, stream>>>(Ab, W1p, b1, Hf, Hb, MTILES);

    // SAGE aggregate: neigh[dst] += h[src]
    k_scatter<false><<<NE / 8, 256, 0, stream>>>(Hf, src, dst, nullptr, neigh, NE);
    // neigh_mean in bf16
    k_scale_cvt<<<(NN * 32 + 255) / 256, 256, 0, stream>>>(neigh, inv_dg, NMb, NN);
    // out = h @ W_self + neigh_mean @ W_neigh + b2
    k_gemm2<<<(MTILES + 7) / 8, 256, 0, stream>>>(Hb, NMb, Wsp, b2, out, MTILES);
}
